// OcclusionDecoder_34454227648763
// MI455X (gfx1250) — compile-verified, matrix operand order checked by reading, not run on hardware
//
#include <hip/hip_runtime.h>
#include <stdint.h>

#define Tn 5
#define Zn 16
#define Yn 96
#define Xn 96
#define Fn 8
#define Cn 128
#define NRAYS 32768
#define KS 128

// Padded bf16 volume: [t' 0..6][y' 0..97][x' 0..97][c 0..127], halo = zeros.
#define PT 7
#define PY 98
#define PX 98
#define NPOS (PT * PY * PX)  // 67228

typedef __attribute__((ext_vector_type(16))) __bf16        v16bf;
typedef __attribute__((ext_vector_type(8)))  float         v8f;
typedef __attribute__((ext_vector_type(4)))  unsigned int  v4u;
typedef __attribute__((ext_vector_type(8)))  int           v8i;
typedef __attribute__((ext_vector_type(4)))  int           v4i;

union Frag {
  v16bf v;
  v4u   u[2];
};

__device__ __forceinline__ unsigned short f2bf(float x) {
  unsigned int b = __float_as_uint(x);
  b += 0x7FFFu + ((b >> 16) & 1u);   // round-to-nearest-even
  return (unsigned short)(b >> 16);
}

// ---------------------------------------------------------------------------
// Kernel A: fold dens_w into conv weights (128 -> 16 output channels) and
// repack into WMMA A-fragment order: wrep[(o*4+kc)*32 + lane][16 halves].
// Also folds conv_b into a per-z bias.
// ---------------------------------------------------------------------------
__global__ void repack_weights(const float* __restrict__ conv_w,
                               const float* __restrict__ conv_b,
                               const float* __restrict__ dens_w,
                               unsigned short* __restrict__ wrep,
                               float* __restrict__ bias) {
  int gid = blockIdx.x * blockDim.x + threadIdx.x;
  if (gid < 16) {
    float acc = 0.f;
    for (int f = 0; f < Fn; ++f) acc += dens_w[f] * conv_b[f * Zn + gid];
    bias[gid] = acc;
  }
  if (gid >= 108 * 512) return;
  int fi = gid >> 9, rem = gid & 511;
  int lane = rem >> 4, h = rem & 15;
  int o = fi >> 2, kc = fi & 3;
  int g = lane >> 4, m = lane & 15;
  int r = h >> 1, pos = h & 1;
  int K  = ((r & 4) ? 16 : 0) + g * 8 + ((r & 3) << 1) + pos;
  int ci = kc * 32 + K;
  float acc = 0.f;
#pragma unroll
  for (int f = 0; f < Fn; ++f)
    acc += dens_w[f] * conv_w[(f * Zn + m) * (Cn * 27) + ci * 27 + o];
  wrep[gid] = f2bf(acc);
}

// ---------------------------------------------------------------------------
// Kernel B: convert f32 volume to zero-padded channel-innermost bf16 buffer.
// One thread writes one 16B octet of channels (c = q*8..q*8+7) at position p.
// ---------------------------------------------------------------------------
__global__ void vol_to_bf16(const float* __restrict__ vol,
                            unsigned short* __restrict__ ibf) {
  int gid = blockIdx.x * blockDim.x + threadIdx.x;
  if (gid >= NPOS * 16) return;
  int p = gid >> 4, q = gid & 15;
  int tp = p / (PY * PX);
  int rest = p % (PY * PX);
  int yp = rest / PX, xp = rest % PX;
  float v[8];
  if (tp >= 1 && tp <= Tn && yp >= 1 && yp <= Yn && xp >= 1 && xp <= Xn) {
    int f = q >> 1, zb = (q & 1) * 8;      // octet stays inside one f-block
    const float* src = vol +
        ((size_t)(f * Tn + (tp - 1)) * Zn + zb) * (Yn * Xn) +
        (size_t)(yp - 1) * Xn + (xp - 1);
#pragma unroll
    for (int j = 0; j < 8; ++j) v[j] = src[(size_t)j * (Yn * Xn)];
  } else {
#pragma unroll
    for (int j = 0; j < 8; ++j) v[j] = 0.f;
  }
  v4u w;
  w.x = (unsigned)f2bf(v[0]) | ((unsigned)f2bf(v[1]) << 16);
  w.y = (unsigned)f2bf(v[2]) | ((unsigned)f2bf(v[3]) << 16);
  w.z = (unsigned)f2bf(v[4]) | ((unsigned)f2bf(v[5]) << 16);
  w.w = (unsigned)f2bf(v[6]) | ((unsigned)f2bf(v[7]) << 16);
  *reinterpret_cast<v4u*>(ibf + (size_t)p * Cn + q * 8) = w;
}

__global__ void init_bounds(unsigned int* __restrict__ bnds) {
  int t = threadIdx.x;
  if (t < Tn) { bnds[2 * t] = 0x7F800000u; bnds[2 * t + 1] = 0u; }
}

// ---------------------------------------------------------------------------
// Kernel C: implicit-GEMM conv via WMMA. One workgroup = one (t,y) row half
// (48 x-positions), 3 waves, each owning a 16-wide N subtile; M = 16 folded
// z-channels; K = 128 cin x 27 taps in chunks of 32.
// The 3x3x50x128ch bf16 halo tile is DMA'd LDS-ward by ONE Tensor Data Mover
// instruction (3-D D#: 9 rows of 12,800B, strides 25,088B / 2,458,624B),
// tracked with TENSORcnt. An asm memory-clobber with the LDS pointer as an
// operand tells the optimizer that the TDM wrote sIn (else the ds reads and
// the LDS allocation get dead-coded, as seen in round 2).
// ---------------------------------------------------------------------------
__global__ void __launch_bounds__(96)
conv_wmma(const unsigned short* __restrict__ ibf,
          const unsigned short* __restrict__ wrep,
          const float* __restrict__ bias,
          float* __restrict__ svol) {
  __shared__ __align__(16) unsigned short sIn[3 * 3 * 50 * Cn];  // 115,200 B

  int b = blockIdx.x;
  int t = b / 192;
  int rest = b % 192;
  int y = rest >> 1;
  int xbase0 = (rest & 1) * 48;

  // Region start in padded buffer: padded coords (t, y, xbase0).
  size_t roff = (((size_t)t * PY + (size_t)y) * PX + (size_t)xbase0) * (Cn * 2);

#if __has_builtin(__builtin_amdgcn_tensor_load_to_lds)
  if (threadIdx.x < 32) {  // one wave issues the TDM op (EXEC ignored by TDM)
    unsigned long long ga = (unsigned long long)(uintptr_t)ibf + roff;
    // ISA 10.2: flat LDS pointer = {shared_aperture, lds_offset}; low 32 bits
    // are the byte offset within this workgroup's LDS allocation.
    unsigned lds_base = (unsigned)(uintptr_t)(void*)sIn;
    // D# group 0: count=1 | lds_addr | global_addr[56:0] | type=2
    v4u g0 = { 1u, lds_base, (unsigned)ga,
               (unsigned)((ga >> 32) & 0x1FFFFFFull) | (2u << 30) };
    // D# group 1: data_size=3 (8B units); tensor_dim0=1600, tensor_dim1=3;
    //             tile_dim0=1600, tile_dim1=3, tile_dim2=3;
    //             stride0=3136 (25,088B), stride1=307,328 (2,458,624B)
    v8i g1 = { (int)(3u << 16),                 // mask=0, data_size=3
               (int)(1600u << 16),              // abar=0 | tdim0.lo16
               (int)(3u << 16),                 // tdim0.hi16=0 | tdim1.lo16
               (int)(1600u << 16),              // tdim1.hi16=0 | tile_dim0
               (int)(3u | (3u << 16)),          // tile_dim1 | tile_dim2
               (int)3136,                       // stride0[31:0]
               (int)0xB0800000u,                // stride0[47:32]=0 | stride1.lo16
               (int)4 };                        // stride1[47:16] = 0x4
    v4i g2 = { 3, 0, 0, 0 };                    // tensor_dim2=3; rest unused
    v4i g3 = { 0, 0, 0, 0 };
#if __clang_major__ >= 23
    v8i g4 = { 0, 0, 0, 0, 0, 0, 0, 0 };
    __builtin_amdgcn_tensor_load_to_lds(g0, g1, g2, g3, g4, 0);
#else
    __builtin_amdgcn_tensor_load_to_lds(g0, g1, g2, g3, 0);
#endif
    __builtin_amdgcn_s_wait_tensorcnt(0);
  }
  // The TDM engine wrote sIn behind the compiler's back: make sIn's address
  // escape into an asm with a memory clobber so the ds reads below survive.
  asm volatile("" : : "v"(sIn) : "memory");
#else
  // Fallback: vectorized copy of the 9 contiguous rows (no converts/branches).
  for (int j = (int)threadIdx.x; j < 7200; j += 96) {   // 7200 x 16B = 115,200B
    int r = j / 800, k = j % 800;
    int tt = r / 3, yy = r % 3;
    const v4u* src = reinterpret_cast<const v4u*>(
        (const char*)ibf + roff + ((size_t)tt * (PY * PX) + (size_t)yy * PX) * (Cn * 2)) + k;
    reinterpret_cast<v4u*>(sIn)[(size_t)r * 800 + k] = *src;
  }
#endif
  __syncthreads();

  int tid = threadIdx.x;
  int wave = tid >> 5;
  int lane = tid & 31;
  int n = lane & 15;
  int g = lane >> 4;

  v8f acc = {0.f, 0.f, 0.f, 0.f, 0.f, 0.f, 0.f, 0.f};

  for (int o = 0; o < 27; ++o) {
    int kt = o / 9, ky = (o % 9) / 3, kx = o % 3;
    int sp = (kt * 3 + ky) * 50 + (wave * 16 + n + kx);  // region spatial idx
    const unsigned short* rowp = &sIn[sp * Cn];
#pragma unroll
    for (int kc = 0; kc < 4; ++kc) {
      Frag bf, af;
      const v4u* bp = reinterpret_cast<const v4u*>(rowp + kc * 32 + g * 8);
      bf.u[0] = bp[0];   // halves K = g*8 .. g*8+7
      bf.u[1] = bp[2];   // halves K = 16+g*8 .. 16+g*8+7
      const v4u* ap =
          reinterpret_cast<const v4u*>(wrep + ((o * 4 + kc) * 32 + lane) * 16);
      af.u[0] = ap[0];
      af.u[1] = ap[1];
      acc = __builtin_amdgcn_wmma_f32_16x16x32_bf16(
          false, af.v, false, bf.v, (short)0, acc, false, false);
    }
  }

  int xg = xbase0 + wave * 16 + n;
#pragma unroll
  for (int r = 0; r < 8; ++r) {
    int m = r + g * 8;  // z index (C-matrix layout: VGPR r, lane group g)
    svol[((t * Zn + m) * Yn + y) * Xn + xg] = acc[r] + bias[m];
  }
}

// ---------------------------------------------------------------------------
// Ray helpers + bounds + march
// ---------------------------------------------------------------------------
__device__ __forceinline__ void ray_nf(const float* o, const float* d,
                                       float& nearv, float& farv) {
  float tmin = -1e30f, tmax = 1e30f;
#pragma unroll
  for (int a = 0; a < 3; ++a) {
    float t1 = (0.f - o[a]) / d[a];
    float t2 = (1.f - o[a]) / d[a];
    tmin = fmaxf(tmin, fminf(t1, t2));
    tmax = fminf(tmax, fmaxf(t1, t2));
  }
  nearv = fmaxf(tmin, 0.01f);
  farv  = fmaxf(tmax, nearv + 1e-6f);
}

__global__ void ray_bounds(const float* __restrict__ ro,
                           const float* __restrict__ rd,
                           unsigned int* __restrict__ bnds) {
  int gid = blockIdx.x * blockDim.x + threadIdx.x;
  if (gid >= Tn * NRAYS) return;
  int t = gid >> 15;
  float o3[3] = {ro[gid * 3 + 0], ro[gid * 3 + 1], ro[gid * 3 + 2]};
  float d3[3] = {rd[gid * 3 + 0], rd[gid * 3 + 1], rd[gid * 3 + 2]};
  float nv, fv;
  ray_nf(o3, d3, nv, fv);
  float dt = (fv - nv) * (1.f / KS);
  float smax = nv + 127.f * dt;
  // starts are always > 0, so float order == u32 bit order
  atomicMin(&bnds[2 * t], __float_as_uint(nv));
  atomicMax(&bnds[2 * t + 1], __float_as_uint(smax));
}

__global__ void render(const float* __restrict__ ro, const float* __restrict__ rd,
                       const float* __restrict__ svol,
                       const float* __restrict__ dens_b,
                       const unsigned int* __restrict__ bnds,
                       float* __restrict__ out) {
  int gid = blockIdx.x * blockDim.x + threadIdx.x;
  if (gid >= Tn * NRAYS) return;
  int t = gid >> 15;
  float ox = ro[gid * 3 + 0], oy = ro[gid * 3 + 1], oz = ro[gid * 3 + 2];
  float dx = rd[gid * 3 + 0], dy = rd[gid * 3 + 1], dz = rd[gid * 3 + 2];
  float o3[3] = {ox, oy, oz}, d3[3] = {dx, dy, dz};
  float nv, fv;
  ray_nf(o3, d3, nv, fv);
  float dt = (fv - nv) * (1.f / KS);
  float db = dens_b[0];
  const float* sv = svol + (size_t)t * Zn * Yn * Xn;

  float cum = 0.f, num = 0.f, den = 0.f;
  for (int i = 0; i < KS; ++i) {
    float start = nv + i * dt;
    float mid = start + 0.5f * dt;
    float px = ox + dx * mid, py = oy + dy * mid, pz = oz + dz * mid;
    float gx = fminf(fmaxf(px, 0.f), 1.f) * (float)(Xn - 1);
    float gy = fminf(fmaxf(py, 0.f), 1.f) * (float)(Yn - 1);
    float gz = fminf(fmaxf(pz, 0.f), 1.f) * (float)(Zn - 1);
    int x0 = (int)gx; int x1 = min(x0 + 1, Xn - 1); float fx = gx - (float)x0;
    int y0 = (int)gy; int y1 = min(y0 + 1, Yn - 1); float fy = gy - (float)y0;
    int z0 = (int)gz; int z1 = min(z0 + 1, Zn - 1); float fz = gz - (float)z0;
    const float* p00 = sv + (z0 * Yn + y0) * Xn;
    const float* p01 = sv + (z0 * Yn + y1) * Xn;
    const float* p10 = sv + (z1 * Yn + y0) * Xn;
    const float* p11 = sv + (z1 * Yn + y1) * Xn;
    float c00 = p00[x0] + fx * (p00[x1] - p00[x0]);
    float c01 = p01[x0] + fx * (p01[x1] - p01[x0]);
    float c10 = p10[x0] + fx * (p10[x1] - p10[x0]);
    float c11 = p11[x0] + fx * (p11[x1] - p11[x0]);
    float c0 = c00 + fy * (c01 - c00);
    float c1 = c10 + fy * (c11 - c10);
    float xin = (c0 + fz * (c1 - c0)) + db;
    float dens = fmaxf(xin, 0.f) + log1pf(expf(-fabsf(xin)));  // softplus
    float dd = dens * dt;
    float Tr = expf(-cum);
    float alpha = 1.f - expf(-dd);
    float w = alpha * Tr;
    num += w * start;
    den += w;
    cum += dd;
  }
  float depth = num / (den + 1e-10f);
  float smin = __uint_as_float(bnds[2 * t]);
  float smax = __uint_as_float(bnds[2 * t + 1]);
  depth = fminf(fmaxf(depth, smin), smax);
  out[gid] = depth;
}

// ---------------------------------------------------------------------------
// d_ws layout:
//   [0,        110592)   repacked bf16 weights (108 frags * 32 lanes * 16 halves)
//   [110592,   110656)   per-z bias (16 f32)
//   [110656,   110696)   per-t start min/max (5 * 2 u32)
//   [114688,   3063808)  scalar field s[t][z][y][x] (f32, 2.95 MB)
//   [3063808, 20274176)  padded bf16 volume ibf[t'][y'][x'][c] (17.2 MB)
// ---------------------------------------------------------------------------
extern "C" void kernel_launch(void* const* d_in, const int* in_sizes, int n_in,
                              void* d_out, int out_size, void* d_ws, size_t ws_size,
                              hipStream_t stream) {
  (void)in_sizes; (void)n_in; (void)out_size; (void)ws_size;
  const float* vol    = (const float*)d_in[0];
  const float* ro     = (const float*)d_in[1];
  const float* rd     = (const float*)d_in[2];
  const float* conv_w = (const float*)d_in[3];
  const float* conv_b = (const float*)d_in[4];
  const float* dens_w = (const float*)d_in[5];
  const float* dens_b = (const float*)d_in[6];
  float* out = (float*)d_out;

  unsigned short* wrep = (unsigned short*)d_ws;
  float* bias          = (float*)((char*)d_ws + 110592);
  unsigned int* bnds   = (unsigned int*)((char*)d_ws + 110656);
  float* svol          = (float*)((char*)d_ws + 114688);
  unsigned short* ibf  = (unsigned short*)((char*)d_ws + 3063808);

  repack_weights<<<216, 256, 0, stream>>>(conv_w, conv_b, dens_w, wrep, bias);
  vol_to_bf16<<<(NPOS * 16 + 255) / 256, 256, 0, stream>>>(vol, ibf);
  init_bounds<<<1, 32, 0, stream>>>(bnds);
  conv_wmma<<<960, 96, 0, stream>>>(ibf, wrep, bias, svol);
  ray_bounds<<<(Tn * NRAYS + 255) / 256, 256, 0, stream>>>(ro, rd, bnds);
  render<<<(Tn * NRAYS + 255) / 256, 256, 0, stream>>>(ro, rd, svol, dens_b, bnds, out);
}
